// QueryPinnedHopfieldLayer_7069516169844
// MI455X (gfx1250) — matrix operand order repaired and checked by reading, not compiled, not run on hardware
//
#include <hip/hip_runtime.h>
#include <hip/hip_bf16.h>
#include <math.h>

// ---------------------------------------------------------------------------
// QueryPinnedHopfieldLayer for MI455X (gfx1250), wave32 + WMMA f16->f32.
// Shapes (fixed by reference setup_inputs):
//   B=8, S=2048, HID=2048, QD=MD=768, N=50000, H=4, D=64, THD=256
// ---------------------------------------------------------------------------

#define B_      8
#define S_      2048
#define HID_    2048
#define QD_     768
#define N_      50000
#define NH_     4
#define HD_     64
#define THD_    256          // NH_*HD_
#define KT_KV   24           // 768 / 32 k-steps for KV projection
#define NPAD_   50016        // N padded to multiple of 32 (1563*32)
#define NKSTEP_ 1563         // NPAD_/32
#define KROWS_  50048        // K16 row allocation (>= 391*128)
#define LN_EPS_ 1e-5f

typedef __attribute__((ext_vector_type(16))) _Float16 v16h;
typedef __attribute__((ext_vector_type(8)))  float    v8f;

// ---------------- WMMA fragment helpers (ISA 7.12.2 layouts, wave32) -------

// 16-bit A matrix 16x32 (MxK), source is row-major tile with leading dim ldm
// (in halfs).  lanes 0-15: M=lane, VGPR j holds K = 2j {+1};
// lanes 16-31: same M, K offset +16 (contiguous K, matches B frags).
__device__ inline v16h load_a_frag(const _Float16* tile, int ldm, int kbase, int lane) {
  union { v16h v; unsigned u[8]; } r;
  const int m    = lane & 15;
  const int koff = (lane & 16) ? 16 : 0;
#pragma unroll
  for (int j = 0; j < 8; ++j) {
    int k = kbase + koff + (j << 1);
    r.u[j] = *(const unsigned*)(tile + (size_t)m * ldm + k);   // two packed halfs
  }
  return r.v;
}

// 16-bit B matrix 32x16 (KxN): lane holds column n = base+(lane&15),
// halfs are 16 consecutive K values starting at kbase + (lane>=16 ? 16 : 0).
// Caller passes a pointer to those 16 consecutive halfs (32B, aligned).
__device__ inline v16h load_b_frag(const _Float16* p) {
  union { v16h v; uint4 q[2]; } u;
  u.q[0] = *(const uint4*)p;          // global_load_b128
  u.q[1] = *(const uint4*)(p + 8);    // global_load_b128
  return u.v;
}

__device__ inline v8f wmma16(v16h a, v16h b, v8f c) {
  return __builtin_amdgcn_wmma_f32_16x16x32_f16(false, a, false, b, (short)0, c,
                                                false, false);
}

// ---------------- kernel 1: pack weights to f16 + zero-init regions --------
// Wpack[kt][c][kk]  kt=0..23, c=0..511 (0..255 -> Wk cols, 256..511 -> Wv),
// kk=0..31.  Lane-contiguous for B fragments.
__global__ void pack_kernel(const float* __restrict__ Wk,
                            const float* __restrict__ Wv,
                            _Float16* __restrict__ wpack,
                            _Float16* __restrict__ vt16,
                            _Float16* __restrict__ w16,
                            float* __restrict__ xi_acc) {
  const long NW   = (long)KT_KV * 512 * 32;                 // Wpack elems
  const long NVP  = (long)NH_ * HD_ * (NPAD_ - N_);         // VT16 pad
  const long NWZ  = (long)NH_ * 16 * NPAD_;                 // weights16 zero
  const long NXA  = (long)NH_ * 8 * HD_;                    // xi_acc zero
  const long TOT  = NW + NVP + NWZ + NXA;
  for (long i = (long)blockIdx.x * blockDim.x + threadIdx.x; i < TOT;
       i += (long)gridDim.x * blockDim.x) {
    if (i < NW) {
      int kt = (int)(i / (512 * 32));
      int c  = (int)((i / 32) % 512);
      int kk = (int)(i % 32);
      float v = (c < THD_) ? Wk[(size_t)c * QD_ + kt * 32 + kk]
                           : Wv[(size_t)(c - THD_) * QD_ + kt * 32 + kk];
      wpack[i] = (_Float16)v;
    } else if (i < NW + NVP) {
      long j = i - NW;                   // (h*64+d)*(pad 16) + p
      long hd = j / (NPAD_ - N_);
      long p  = j % (NPAD_ - N_);
      vt16[hd * NPAD_ + N_ + p] = (_Float16)0.0f;
    } else if (i < NW + NVP + NWZ) {
      w16[i - NW - NVP] = (_Float16)0.0f;
    } else {
      xi_acc[i - NW - NVP - NWZ] = 0.0f;
    }
  }
}

// ---------------- kernel 2: query layernorm + projection -> xi16 -----------
// xi16 layout: [h][16][64] f16 (rows 8..15 zero-padded for WMMA A tiles)
__global__ void qinit_kernel(const float* __restrict__ qe,
                             const float* __restrict__ Wq,
                             const float* __restrict__ gq,
                             const float* __restrict__ bq,
                             _Float16* __restrict__ xi16) {
  __shared__ float qn[B_][QD_];
  __shared__ float rs[256], rq[256];
  __shared__ float mu_s[B_], rstd_s[B_];
  const int t = threadIdx.x, row = t >> 5, l = t & 31;
  float s = 0.f, sq = 0.f;
  for (int j = l; j < QD_; j += 32) {
    float x = qe[row * QD_ + j]; s += x; sq += x * x;
  }
  rs[t] = s; rq[t] = sq; __syncthreads();
  for (int o = 16; o > 0; o >>= 1) {
    if (l < o) { rs[t] += rs[t + o]; rq[t] += rq[t + o]; }
    __syncthreads();
  }
  if (l == 0) {
    float mu = rs[t] / QD_;
    float var = rq[t] / QD_ - mu * mu;
    mu_s[row] = mu; rstd_s[row] = rsqrtf(var + LN_EPS_);
  }
  __syncthreads();
  for (int j = l; j < QD_; j += 32) {
    float x = qe[row * QD_ + j];
    qn[row][j] = (x - mu_s[row]) * rstd_s[row] * gq[j] + bq[j];
  }
  __syncthreads();
  for (int o = t; o < B_ * THD_; o += 256) {
    int b = o >> 8, c = o & 255;
    float acc = 0.f;
    for (int k = 0; k < QD_; ++k) acc += qn[b][k] * Wq[(size_t)c * QD_ + k];
    int h = c >> 6, d = c & 63;
    xi16[((size_t)h * 16 + b) * HD_ + d] = (_Float16)acc;
  }
  for (int o = t; o < NH_ * 8 * HD_; o += 256) {   // zero pad rows 8..15
    int h = o / (8 * HD_), r = 8 + (o / HD_) % 8, d = o % HD_;
    xi16[((size_t)h * 16 + r) * HD_ + d] = (_Float16)0.0f;
  }
}

// ---------------- kernel 3: LN(memory_bank) + K/V projection (WMMA) --------
// One block = 16 memory rows; 8 waves, each owns 4 of 32 output col-tiles.
// K16:  [h][KROWS_][64]  (row-contiguous in d -> B frags for scores)
// VT16: [h][64][NPAD_]   (row-contiguous in n -> B frags for aggregation)
__global__ void kv_kernel(const float* __restrict__ mb,
                          const float* __restrict__ gm,
                          const float* __restrict__ bm,
                          const _Float16* __restrict__ wpack,
                          _Float16* __restrict__ k16,
                          _Float16* __restrict__ vt16) {
  __shared__ _Float16 at[16 * 776];     // 16 x 768 f16 tile (ld=776)
  __shared__ float rs[256], rq[256];
  __shared__ float mu_s[16], rstd_s[16];
  const int t = threadIdx.x;
  const int n0 = blockIdx.x * 16;

  { // layernorm: 16 threads per row
    int r = t >> 4, l16 = t & 15;
    const float* src = mb + (size_t)(n0 + r) * QD_;
    float s = 0.f, sq = 0.f;
    for (int j = l16; j < QD_; j += 16) { float x = src[j]; s += x; sq += x * x; }
    rs[t] = s; rq[t] = sq; __syncthreads();
    if (l16 == 0) {
      float ss = 0.f, qq = 0.f;
      for (int i = 0; i < 16; ++i) { ss += rs[t + i]; qq += rq[t + i]; }
      float mu = ss / QD_;
      mu_s[r] = mu; rstd_s[r] = rsqrtf(qq / QD_ - mu * mu + LN_EPS_);
    }
    __syncthreads();
    for (int j = l16; j < QD_; j += 16) {
      float x = src[j];
      at[r * 776 + j] = (_Float16)((x - mu_s[r]) * rstd_s[r] * gm[j] + bm[j]);
    }
  }
  __syncthreads();

  const int wave = t >> 5, lane = t & 31;
  const int cbase = wave * 64 + (lane & 15);            // col of tile i = cbase+16i
  const _Float16* bsel = wpack + ((lane & 16) ? 16 : 0);
  v8f acc[4] = {};
#pragma unroll 1
  for (int kt = 0; kt < KT_KV; ++kt) {
    // group all loads of this k-step first (one clause), then 4 WMMAs
    v16h a = load_a_frag(at, 776, kt * 32, lane);
    v16h bf[4];
#pragma unroll
    for (int i = 0; i < 4; ++i)
      bf[i] = load_b_frag(bsel + ((size_t)(kt * 512 + cbase + i * 16)) * 32);
#pragma unroll
    for (int i = 0; i < 4; ++i) acc[i] = wmma16(a, bf[i], acc[i]);
  }
#pragma unroll
  for (int i = 0; i < 4; ++i) {
    int c = cbase + i * 16;
    if (c < THD_) {              // K output (scatter, d-major rows)
      int h = c >> 6, d = c & 63;
#pragma unroll
      for (int v = 0; v < 8; ++v) {
        int n = n0 + v + ((lane & 16) ? 8 : 0);
        k16[((size_t)h * KROWS_ + n) * HD_ + d] = (_Float16)acc[i][v];
      }
    } else {                     // V output (transposed layout, vectorized b128)
      int cv = c - THD_, h = cv >> 6, d = cv & 63;
      size_t base = ((size_t)h * HD_ + d) * NPAD_ + n0 + ((lane & 16) ? 8 : 0);
      union { uint4 q; _Float16 hl[8]; } pk;
#pragma unroll
      for (int v = 0; v < 8; ++v) pk.hl[v] = (_Float16)acc[i][v];
      *(uint4*)(vt16 + base) = pk.q;
    }
  }
}

// ---------------- kernel 4: scores = beta_h * xi . K  (WMMA) ---------------
// grid (ceil(N/128), NH_); 8 waves x 16-col n-tiles.  scores: [h][8][N] f32
__global__ void scores_kernel(const _Float16* __restrict__ xi16,
                              const _Float16* __restrict__ k16,
                              const float* __restrict__ log_beta,
                              float* __restrict__ scores) {
  const int h = blockIdx.y;
  const int wave = threadIdx.x >> 5, lane = threadIdx.x & 31;
  const int nbase = blockIdx.x * 128 + wave * 16;
  if (nbase >= N_) return;
  const float beta = __expf(log_beta[h]);
  const _Float16* atile = xi16 + (size_t)h * 16 * HD_;
  const _Float16* bp = k16 + ((size_t)h * KROWS_ + nbase + (lane & 15)) * HD_
                       + ((lane & 16) ? 16 : 0);
  v16h a0 = load_a_frag(atile, HD_, 0, lane);
  v16h a1 = load_a_frag(atile, HD_, 32, lane);
  v16h b0 = load_b_frag(bp);
  v16h b1 = load_b_frag(bp + 32);
  v8f acc = {};
  acc = wmma16(a0, b0, acc);
  acc = wmma16(a1, b1, acc);
  if (lane < 16) {                            // rows m=0..7 == batch ids
    int n = nbase + lane;
#pragma unroll
    for (int v = 0; v < 8; ++v)
      scores[((size_t)h * B_ + v) * N_ + n] = acc[v] * beta;
  }
}

// ---------------- kernel 5: entmax-1.5 via bisection (LDS-resident) --------
// grid (B_, NH_), 512 threads, dynamic LDS = N_*4 bytes.
// Row staged to LDS via gfx1250 async direct-to-LDS copies (ASYNCcnt).
// weights16: [h][16][NPAD_] f16 (rows 8..15 and cols >= N_ stay zero)
__global__ void entmax_kernel(const float* __restrict__ scores,
                              _Float16* __restrict__ w16) {
  extern __shared__ float z[];          // 50000 floats = 200 KB
  __shared__ float red[512];
  const int b = blockIdx.x, h = blockIdx.y, t = threadIdx.x;
  const float* row = scores + ((size_t)h * B_ + b) * N_;

  { // bulk: 24 iters x 512 lanes x 16B async = 196608 B; scalar tail
    __attribute__((address_space(3))) float* zl =
        (__attribute__((address_space(3))) float*)z;
#pragma unroll 1
    for (int it = 0; it < 24; ++it) {
      int f0 = (it * 512 + t) * 4;
      unsigned lofs = (unsigned)(size_t)(zl + f0);
      unsigned long long ga = (unsigned long long)(row + f0);
      asm volatile("global_load_async_to_lds_b128 %0, %1, off"
                   :: "v"(lofs), "v"(ga) : "memory");
    }
    for (int n = 24 * 2048 + t; n < N_; n += 512) z[n] = row[n];
    asm volatile("s_wait_asynccnt 0" ::: "memory");
  }
  __syncthreads();

  float lmax = -3.4e38f;
  for (int n = t; n < N_; n += 512) lmax = fmaxf(lmax, z[n]);
  red[t] = lmax; __syncthreads();
  for (int o = 256; o > 0; o >>= 1) {
    if (t < o) red[t] = fmaxf(red[t], red[t + o]);
    __syncthreads();
  }
  const float zmax = red[0]; __syncthreads();
  for (int n = t; n < N_; n += 512) z[n] = (z[n] - zmax) * 0.5f;
  __syncthreads();

  // f(tau) = sum relu(z - tau)^2 is monotone decreasing; f(-1)>=1, f(0)=0.
  float lo = -1.0f, hi = 0.0f;
  for (int it = 0; it < 30; ++it) {
    float tau = 0.5f * (lo + hi), part = 0.f;
    for (int n = t; n < N_; n += 512) {
      float d = z[n] - tau;
      if (d > 0.f) part += d * d;
    }
    red[t] = part; __syncthreads();
    for (int o = 256; o > 0; o >>= 1) {
      if (t < o) red[t] += red[t + o];
      __syncthreads();
    }
    float f = red[0]; __syncthreads();
    if (f >= 1.0f) lo = tau; else hi = tau;
  }
  const float tau = 0.5f * (lo + hi);
  _Float16* out = w16 + ((size_t)h * 16 + b) * NPAD_;
  for (int n = t; n < NPAD_; n += 512) {
    float w = 0.f;
    if (n < N_) { float d = z[n] - tau; if (d > 0.f) w = d * d; }
    out[n] = (_Float16)w;
  }
}

// ---------------- kernel 6: xi = weights @ V  (WMMA + f32 atomics) ---------
// grid (KSLICES, NH_), 128 threads (4 waves; wave = d-tile).
#define KSLICES_ 28
__global__ void agg_kernel(const _Float16* __restrict__ w16,
                           const _Float16* __restrict__ vt16,
                           float* __restrict__ xi_acc) {
  const int h = blockIdx.y, slice = blockIdx.x;
  const int wave = threadIdx.x >> 5, lane = threadIdx.x & 31;
  const int ks0 = slice * 56;
  const int ks1 = min(ks0 + 56, NKSTEP_);
  if (ks0 >= ks1) return;
  const _Float16* atile = w16 + (size_t)h * 16 * NPAD_;
  const int d = wave * 16 + (lane & 15);
  const _Float16* bcol = vt16 + ((size_t)h * HD_ + d) * NPAD_ + ((lane & 16) ? 16 : 0);
  v8f acc = {};
  for (int ks = ks0; ks < ks1; ++ks) {
    __builtin_prefetch(bcol + (ks + 2) * 32, 0, 0);   // global_prefetch_b8
    v16h a = load_a_frag(atile, NPAD_, ks * 32, lane);
    acc = wmma16(a, load_b_frag(bcol + ks * 32), acc);
  }
  if (lane < 16) {
    int dd = wave * 16 + lane;
#pragma unroll
    for (int v = 0; v < 8; ++v)
      atomicAdd(&xi_acc[((size_t)h * B_ + v) * HD_ + dd], acc[v]);
  }
}

// ---------------- kernel 7: xi_acc -> xi16 (+xi32 copy), re-zero acc -------
__global__ void xicvt_kernel(float* __restrict__ xi_acc,
                             _Float16* __restrict__ xi16,
                             float* __restrict__ xi32) {
  for (int i = threadIdx.x; i < NH_ * 16 * HD_; i += 256) {
    int h = i >> 10, r = (i >> 6) & 15, d = i & 63;
    float v = 0.f;
    if (r < 8) {
      size_t src = ((size_t)h * B_ + r) * HD_ + d;
      v = xi_acc[src];
      xi32[src] = v;
      xi_acc[src] = 0.f;
    }
    xi16[i] = (_Float16)v;
  }
}

// ---------------- kernel 8: out-row = retrieved @ Wo^T ---------------------
__global__ void outproj_kernel(const float* __restrict__ xi32,
                               const float* __restrict__ Wo,
                               float* __restrict__ o32) {
  __shared__ float r[THD_];
  const int b = blockIdx.y, t = threadIdx.x;
  r[t] = xi32[((size_t)(t >> 6) * B_ + b) * HD_ + (t & 63)];   // c = h*64+d
  __syncthreads();
  int j = blockIdx.x * 256 + t;
  float acc = 0.f;
  for (int c = 0; c < THD_; ++c) acc += r[c] * Wo[(size_t)j * THD_ + c];
  o32[(size_t)b * HID_ + j] = acc;
}

// ---------------- kernel 9: broadcast over S (bandwidth kernel) ------------
__global__ void bcast_kernel(const float* __restrict__ o32,
                             float* __restrict__ out) {
  const int bx = blockIdx.x;         // b*S + s
  const int b = bx >> 11;
  const float4* src = (const float4*)(o32 + (size_t)b * HID_);
  float4* dst = (float4*)(out + (size_t)bx * HID_);
  for (int i = threadIdx.x; i < HID_ / 4; i += 256) dst[i] = src[i];
}

// ---------------------------------------------------------------------------
extern "C" void kernel_launch(void* const* d_in, const int* in_sizes, int n_in,
                              void* d_out, int out_size, void* d_ws, size_t ws_size,
                              hipStream_t stream) {
  const float* qe  = (const float*)d_in[1];
  const float* mb  = (const float*)d_in[2];
  const float* Wq  = (const float*)d_in[3];
  const float* Wk  = (const float*)d_in[4];
  const float* Wv  = (const float*)d_in[5];
  const float* Wo  = (const float*)d_in[6];
  const float* lb  = (const float*)d_in[7];
  const float* gq  = (const float*)d_in[8];
  const float* bq  = (const float*)d_in[9];
  const float* gm  = (const float*)d_in[10];
  const float* bm  = (const float*)d_in[11];
  float* out = (float*)d_out;

  size_t off = 0;
  auto alloc = [&](size_t bytes) {
    void* p = (char*)d_ws + off;
    off = (off + bytes + 255) & ~(size_t)255;
    return p;
  };
  _Float16* wpack = (_Float16*)alloc((size_t)KT_KV * 512 * 32 * 2);
  _Float16* k16   = (_Float16*)alloc((size_t)NH_ * KROWS_ * HD_ * 2);
  _Float16* vt16  = (_Float16*)alloc((size_t)NH_ * HD_ * NPAD_ * 2);
  float*    sc    = (float*)   alloc((size_t)NH_ * B_ * N_ * 4);
  _Float16* w16   = (_Float16*)alloc((size_t)NH_ * 16 * NPAD_ * 2);
  _Float16* xi16  = (_Float16*)alloc((size_t)NH_ * 16 * HD_ * 2);
  float*    xiacc = (float*)   alloc((size_t)NH_ * B_ * HD_ * 4);
  float*    xi32  = (float*)   alloc((size_t)NH_ * B_ * HD_ * 4);
  float*    o32   = (float*)   alloc((size_t)B_ * HID_ * 4);
  (void)ws_size; (void)in_sizes; (void)n_in; (void)out_size;

  pack_kernel<<<2048, 256, 0, stream>>>(Wk, Wv, wpack, vt16, w16, xiacc);
  qinit_kernel<<<1, 256, 0, stream>>>(qe, Wq, gq, bq, xi16);
  kv_kernel<<<N_ / 16, 256, 0, stream>>>(mb, gm, bm, wpack, k16, vt16);

  for (int step = 0; step < 3; ++step) {
    scores_kernel<<<dim3((N_ + 127) / 128, NH_), 256, 0, stream>>>(xi16, k16, lb, sc);
    entmax_kernel<<<dim3(B_, NH_), 512, N_ * sizeof(float), stream>>>(sc, w16);
    agg_kernel<<<dim3(KSLICES_, NH_), 128, 0, stream>>>(w16, vt16, xiacc);
    xicvt_kernel<<<1, 256, 0, stream>>>(xiacc, xi16, xi32);
  }

  outproj_kernel<<<dim3(HID_ / 256, B_), 256, 0, stream>>>(xi32, Wo, o32);
  bcast_kernel<<<B_ * S_, 256, 0, stream>>>(o32, out);
}
